// Circle_Rank_loss_41678362640825
// MI455X (gfx1250) — matrix operand order repaired and checked by reading, not compiled, not run on hardware
//
#include <hip/hip_runtime.h>
#include <hip/hip_bf16.h>
#include <math.h>

#define NN 4096
#define DD 256

#define M1C 2.0f
#define M2C 2.0f
#define A1C 2.4f
#define A2C 2.2f
#define EPSC 1e-5f

// LDS B-panel: 128 columns x 256 K bf16, padded 16B per 512B row by the TDM
// => column stride 528 bytes (264 bf16 elements). Lane l (0..15) reads at
// byte offset 528*l + 32*k -> bank quadruples 4l..4l+3: conflict-free.
#define COLS_PER_BLOCK 128
#define COL_STRIDE_ELEM 264
#define PANEL_BYTES (COLS_PER_BLOCK * COL_STRIDE_ELEM * 2)  // 67584

typedef __attribute__((ext_vector_type(16))) __bf16 v16bf;
typedef __attribute__((ext_vector_type(8)))  __bf16 v8bf;
typedef __attribute__((ext_vector_type(8)))  float  v8f;
typedef __attribute__((ext_vector_type(4)))  unsigned int u32x4;
typedef __attribute__((ext_vector_type(8)))  int i32x8;
typedef __attribute__((ext_vector_type(4)))  int i32x4;

#if __has_include(<hip/amd_detail/amd_gfx1250_TDM.h>)
#define TDM_SIX_ARGS 1
#else
#define TDM_SIX_ARGS 0
#endif

// ---------------------------------------------------------------------------
// Issue one TDM load: 2D tile of 128 rows x 256 bf16 (contiguous panel of the
// row-major xn matrix) into LDS, with 4-DWORD padding every 128 DWORDs.
// D# layout per CDNA5 ISA.
// ---------------------------------------------------------------------------
__device__ inline void tdm_load_panel(const __bf16* gptr, unsigned lds_off) {
  const unsigned long long ga = (unsigned long long)(uintptr_t)gptr;
  u32x4 g0;
  g0[0] = 1u;                                   // count=1, user descriptor
  g0[1] = lds_off;                              // lds_addr (bytes)
  g0[2] = (unsigned)ga;                         // global_addr[31:0]
  g0[3] = (unsigned)((ga >> 32) & 0x01FFFFFFull) | (2u << 30);  // addr[56:32] | type=2
  i32x8 g1;
  g1[0] = (int)((1u << 16)      // data_size = 2 bytes
              | (1u << 20)      // pad_enable
              | (6u << 22)      // pad_interval: every 128 DWORDs (512B)
              | (3u << 25));    // pad_amount: 4 DWORDs (16B)
  g1[1] = (int)(256u << 16);    // tensor_dim0 = 256 (bits 79:48, low half)
  g1[2] = (int)(4096u << 16);   // tensor_dim0 hi=0 | tensor_dim1 = 4096 (low)
  g1[3] = (int)(256u << 16);    // tensor_dim1 hi=0 | tile_dim0 = 256
  g1[4] = COLS_PER_BLOCK;       // tile_dim1 = 128 | tile_dim2 = 0
  g1[5] = 256;                  // tensor_dim0_stride = 256 (low 32)
  g1[6] = 0;                    // stride hi | tensor_dim1_stride low
  g1[7] = 0;
  const i32x4 z4 = {0, 0, 0, 0};
#if TDM_SIX_ARGS
  const i32x8 z8 = {0, 0, 0, 0, 0, 0, 0, 0};
  __builtin_amdgcn_tensor_load_to_lds(g0, g1, z4, z4, z8, 0);
#else
  __builtin_amdgcn_tensor_load_to_lds(g0, g1, z4, z4, 0);
#endif
}

// ---------------------------------------------------------------------------
// A-fragment (16-bit A, 16x32): lanes hold K = k0..k0+15 contiguous,
// k0 = (lane>=16 ? 16 : 0). Two 16B loads at +0 and +8 elements.
// ---------------------------------------------------------------------------
__device__ inline v16bf load_fragA(const __bf16* base) {
  v8bf a = *reinterpret_cast<const v8bf*>(base);
  v8bf b = *reinterpret_cast<const v8bf*>(base + 8);
  v16bf r;
#pragma unroll
  for (int i = 0; i < 8; ++i) { r[i] = a[i]; r[i + 8] = b[i]; }
  return r;
}

// ---------------------------------------------------------------------------
// B-fragment (16-bit B, 32x16): lane&15 = N, low lanes K = kb..kb+15, high
// lanes K = kb+16..kb+31 -> 32 contiguous bytes (two ds_load_b128 from LDS).
// ---------------------------------------------------------------------------
__device__ inline v16bf load_fragB(const __bf16* base) {
  v8bf a = *reinterpret_cast<const v8bf*>(base);
  v8bf b = *reinterpret_cast<const v8bf*>(base + 8);
  v16bf r;
#pragma unroll
  for (int i = 0; i < 8; ++i) { r[i] = a[i]; r[i + 8] = b[i]; }
  return r;
}

// ---------------------------------------------------------------------------
// Kernel A: row-normalize x, emit bf16 hi/lo split + row squared-norm.
// ---------------------------------------------------------------------------
__global__ __launch_bounds__(DD) void normalize_split_kernel(
    const float* __restrict__ x, __bf16* __restrict__ xhi,
    __bf16* __restrict__ xlo, float* __restrict__ sq) {
  const int row = blockIdx.x;
  const int t = threadIdx.x;
  float v = x[row * DD + t];
  float ss = v * v;
#pragma unroll
  for (int m = 16; m > 0; m >>= 1) ss += __shfl_xor(ss, m, 32);
  __shared__ float wred[DD / 32];
  if ((t & 31) == 0) wred[t >> 5] = ss;
  __syncthreads();
  float tot = 0.0f;
#pragma unroll
  for (int k = 0; k < DD / 32; ++k) tot += wred[k];
  const float nrm = sqrtf(tot);
  const float inv = 1.0f / fmaxf(nrm, 1e-12f);
  const float xn = v * inv;
  const __bf16 h = (__bf16)xn;
  xhi[row * DD + t] = h;
  xlo[row * DD + t] = (__bf16)(xn - (float)h);
  if (t == 0) sq[row] = tot * inv * inv;
}

// ---------------------------------------------------------------------------
// Zero the per-row loss accumulators (workspace is poisoned by harness).
// ---------------------------------------------------------------------------
__global__ void zero_kernel(float* __restrict__ p, int n) {
  int i = blockIdx.x * blockDim.x + threadIdx.x;
  if (i < n) p[i] = 0.0f;
}

// ---------------------------------------------------------------------------
// Positive-pair counts are pure functions of targets/sub: compute them in a
// cheap standalone pass so the hot kernel carries 6 (not 8) accumulator sets.
// ---------------------------------------------------------------------------
__global__ __launch_bounds__(256) void count_kernel(
    const int* __restrict__ t, const int* __restrict__ s,
    float* __restrict__ rowacc) {
  const int i = blockIdx.x;
  const int ti = t[i], si = s[i];
  int ci = 0, cc = 0;
  for (int j = threadIdx.x; j < NN; j += 256) {
    if (t[j] == ti && j != i) {
      if (s[j] == si) ci++; else cc++;
    }
  }
  __shared__ int r0[256], r1[256];
  r0[threadIdx.x] = ci;
  r1[threadIdx.x] = cc;
  __syncthreads();
  for (int off = 128; off > 0; off >>= 1) {
    if (threadIdx.x < off) {
      r0[threadIdx.x] += r0[threadIdx.x + off];
      r1[threadIdx.x] += r1[threadIdx.x + off];
    }
    __syncthreads();
  }
  if (threadIdx.x == 0) {
    rowacc[1 * NN + i] = (float)r0[0];
    rowacc[3 * NN + i] = (float)r1[0];
  }
}

// ---------------------------------------------------------------------------
// Kernel B: fused Gram GEMM (split-bf16 WMMA) + dist store + loss epilogue.
// Block = 256 threads = 8 waves owning a 128-row strip; grid.y picks a
// 128-column chunk whose hi/lo B panels are staged in LDS by the TDM and
// shared by all 8 waves.
// ---------------------------------------------------------------------------
__global__ __launch_bounds__(256) void gram_loss_kernel(
    const __bf16* __restrict__ xhi, const __bf16* __restrict__ xlo,
    const float* __restrict__ sq, const int* __restrict__ targets,
    const int* __restrict__ sub, float* __restrict__ dist,
    float* __restrict__ rowacc) {
  extern __shared__ __align__(16) char smem[];
  __bf16* bhi_l = (__bf16*)smem;
  __bf16* blo_l = (__bf16*)(smem + PANEL_BYTES);

  const int lane = threadIdx.x & 31;
  const int wave = threadIdx.x >> 5;
  const int l15 = lane & 15;
  const int hi16 = lane >> 4;  // 0 or 1: which half of the wave

  const int row_base = blockIdx.x * 128 + wave * 16;
  const int col_chunk = blockIdx.y * COLS_PER_BLOCK;

  // One wave kicks off the async TDM panel loads (EXEC-independent DMA).
  if (threadIdx.x < 32) {
    tdm_load_panel(xhi + (size_t)col_chunk * DD, (unsigned)(uintptr_t)bhi_l);
    tdm_load_panel(xlo + (size_t)col_chunk * DD, (unsigned)(uintptr_t)blo_l);
  }

  // Overlap with the TDM: per-lane row attributes + A fragments (registers).
  int   t_i[8], s_i[8];
  float q_i[8];
#pragma unroll
  for (int r = 0; r < 8; ++r) {
    const int i = row_base + r + 8 * hi16;
    t_i[r] = targets[i];
    s_i[r] = sub[i];
    q_i[r] = sq[i];
  }
  const int am  = row_base + l15;
  const int ak0 = hi16 * 16;
  v16bf ahi[8], alo[8];
#pragma unroll
  for (int kc = 0; kc < 8; ++kc) {
    ahi[kc] = load_fragA(xhi + am * DD + kc * 32 + ak0);
    alo[kc] = load_fragA(xlo + am * DD + kc * 32 + ak0);
  }

  if (threadIdx.x < 32) __builtin_amdgcn_s_wait_tensorcnt(0);
  __syncthreads();  // panels visible to all 8 waves

  // 6 quantities x 8 C-rows per lane:
  // 0 ap_i, 1 ap_c, 2 num_i, 3 wsum_i, 4 num_c, 5 wsum_c
  float acc[8][6];
#pragma unroll
  for (int r = 0; r < 8; ++r)
#pragma unroll
    for (int q = 0; q < 6; ++q) acc[r][q] = 0.0f;

  for (int ct = 0; ct < COLS_PER_BLOCK / 16; ++ct) {
    const int col_local = ct * 16 + l15;   // this lane's column in the panel
    const int j = col_chunk + col_local;   // global column
    const int boff = col_local * COL_STRIDE_ELEM + hi16 * 16;

    v8f c;
#pragma unroll
    for (int e = 0; e < 8; ++e) c[e] = 0.0f;

#pragma unroll
    for (int kc = 0; kc < 8; ++kc) {
      const v16bf bhi = load_fragB(bhi_l + boff + kc * 32);
      const v16bf blo = load_fragB(blo_l + boff + kc * 32);
      // dot(f32) ~= hi*hi + hi*lo + lo*hi (lo*lo dropped, ~2^-16 rel. err.)
      c = __builtin_amdgcn_wmma_f32_16x16x32_bf16(false, ahi[kc], false, bhi,
                                                  (short)0, c, false, false);
      c = __builtin_amdgcn_wmma_f32_16x16x32_bf16(false, ahi[kc], false, blo,
                                                  (short)0, c, false, false);
      c = __builtin_amdgcn_wmma_f32_16x16x32_bf16(false, alo[kc], false, bhi,
                                                  (short)0, c, false, false);
    }

    const int   tj = targets[j];
    const int   sj = sub[j];
    const float qj = sq[j];

#pragma unroll
    for (int r = 0; r < 8; ++r) {
      const int i = row_base + r + 8 * hi16;
      const float d2   = q_i[r] + qj - 2.0f * c[r];
      const float dval = sqrtf(fmaxf(d2, 1e-12f));
      dist[(size_t)i * NN + j] = dval;

      const bool same_t = (t_i[r] == tj);
      const bool eye    = (i == j);
      const bool intra  = (s_i[r] == sj);
      const bool pos    = same_t && !eye;
      const bool neg    = !same_t;

      acc[r][0] += (pos && intra)  ? fmaxf(dval + (M1C - A1C), 0.0f) : 0.0f;
      acc[r][1] += (pos && !intra) ? fmaxf(dval + (M2C - A2C), 0.0f) : 0.0f;

      const bool sel_i = neg && intra && (dval < A1C);
      const bool sel_c = neg && !intra && (dval < A2C);
      const float wi = sel_i ? __expf(A1C - dval) : 0.0f;
      const float wc = sel_c ? __expf(A2C - dval) : 0.0f;
      acc[r][2] += fmaxf((A1C - dval) * wi, 0.0f);
      acc[r][3] += wi;
      acc[r][4] += (A2C - dval) * wc;
      acc[r][5] += wc;
    }
  }

  // Cross-lane reduction over the 16 lanes of each half (same C-matrix row).
  const int qmap[6] = {0, 2, 4, 5, 6, 7};
#pragma unroll
  for (int r = 0; r < 8; ++r) {
#pragma unroll
    for (int q = 0; q < 6; ++q) {
      float v = acc[r][q];
      v += __shfl_xor(v, 1, 32);
      v += __shfl_xor(v, 2, 32);
      v += __shfl_xor(v, 4, 32);
      v += __shfl_xor(v, 8, 32);
      if (l15 == 0) {
        const int i = row_base + r + 8 * hi16;
        atomicAdd(&rowacc[qmap[q] * NN + i], v);
      }
    }
  }
}

// ---------------------------------------------------------------------------
// Kernel C: combine per-row accumulators into the scalar mean loss.
// ---------------------------------------------------------------------------
__global__ __launch_bounds__(256) void finalize_kernel(
    const float* __restrict__ rowacc, float* __restrict__ out) {
  __shared__ float red[256];
  float s = 0.0f;
  for (int i = threadIdx.x; i < NN; i += 256) {
    const float api = rowacc[0 * NN + i];
    const float ci  = rowacc[1 * NN + i];
    const float apc = rowacc[2 * NN + i];
    const float cc  = rowacc[3 * NN + i];
    const float ni  = rowacc[4 * NN + i];
    const float wi  = rowacc[5 * NN + i];
    const float nc  = rowacc[6 * NN + i];
    const float wc  = rowacc[7 * NN + i];
    s += api / (ci + EPSC) + apc / (cc + EPSC) + ni / (wi + EPSC) +
         nc / (wc + EPSC);
  }
  red[threadIdx.x] = s;
  __syncthreads();
  for (int off = 128; off > 0; off >>= 1) {
    if (threadIdx.x < off) red[threadIdx.x] += red[threadIdx.x + off];
    __syncthreads();
  }
  if (threadIdx.x == 0) out[0] = red[0] / (float)NN;
}

// ---------------------------------------------------------------------------
extern "C" void kernel_launch(void* const* d_in, const int* in_sizes, int n_in,
                              void* d_out, int out_size, void* d_ws,
                              size_t ws_size, hipStream_t stream) {
  const float* x       = (const float*)d_in[0];
  const int*   targets = (const int*)d_in[1];
  const int*   sub     = (const int*)d_in[2];

  float* out   = (float*)d_out;   // out[0] = loss, out[1..] = dist (4096x4096)
  float* distm = out + 1;

  char* ws = (char*)d_ws;
  __bf16* xhi    = (__bf16*)ws;                                  // 2 MB
  __bf16* xlo    = (__bf16*)(ws + (size_t)NN * DD * 2);          // 2 MB
  float*  sq     = (float*)(ws + (size_t)NN * DD * 4);           // 16 KB
  float*  rowacc = (float*)(ws + (size_t)NN * DD * 4 + NN * 4);  // 128 KB

  static int shmem_set = 0;
  (void)shmem_set;
  hipFuncSetAttribute((const void*)gram_loss_kernel,
                      hipFuncAttributeMaxDynamicSharedMemorySize,
                      2 * PANEL_BYTES);

  normalize_split_kernel<<<NN, DD, 0, stream>>>(x, xhi, xlo, sq);
  zero_kernel<<<(8 * NN + 255) / 256, 256, 0, stream>>>(rowacc, 8 * NN);
  count_kernel<<<NN, 256, 0, stream>>>(targets, sub, rowacc);

  dim3 grid(NN / 128, NN / COLS_PER_BLOCK);  // 32 x 32
  gram_loss_kernel<<<grid, 256, 2 * PANEL_BYTES, stream>>>(
      xhi, xlo, sq, targets, sub, distm, rowacc);

  finalize_kernel<<<1, 256, 0, stream>>>(rowacc, out);
}
